// MOELayer_19327352832435
// MI455X (gfx1250) — matrix operand order repaired and checked by reading, not compiled, not run on hardware
//
#include <hip/hip_runtime.h>
#include <hip/hip_bf16.h>
#include <stdint.h>

#define D_MODEL 1024
#define D_FF    4096
#define N_EXP   8
#define TOPK    2
#define T_TOK   4096
#define R_ASN   (T_TOK * TOPK)

typedef __attribute__((ext_vector_type(16))) __bf16 v16bf;
typedef __attribute__((ext_vector_type(8)))  float  v8f;

struct BfFrag { uint4 lo, hi; };   // 32 bytes = 16 bf16 = one 16x32 fragment slice per lane

__device__ __forceinline__ unsigned short f2bf(float f) {
  unsigned u = __float_as_uint(f);
  unsigned r = u + 0x7FFFu + ((u >> 16) & 1u);   // round-to-nearest-even
  return (unsigned short)(r >> 16);
}

// ---- CDNA5 async global->LDS staging (guarded; falls back to load+ds_store) ----
#if defined(__has_builtin)
#  if __has_builtin(__builtin_amdgcn_global_load_async_to_lds_b128) && \
      __has_builtin(__builtin_amdgcn_s_wait_asynccnt)
#    define USE_ASYNC_LDS 1
#  endif
#endif
#ifndef USE_ASYNC_LDS
#  define USE_ASYNC_LDS 0
#endif

#if USE_ASYNC_LDS
// builtin signature (from hipcc diagnostic): arg0 = int4 (vector_size 16) pointer in AS1
typedef int i4v __attribute__((vector_size(16)));
typedef __attribute__((address_space(1))) i4v as1_i4v;   // global
typedef __attribute__((address_space(3))) i4v as3_i4v;   // LDS
#endif

// stage one 64B row slice of the B tile (per thread)
__device__ __forceinline__ void stage_b(const unsigned char* src, unsigned char* dstLds) {
#if USE_ASYNC_LDS
  __builtin_amdgcn_global_load_async_to_lds_b128((as1_i4v*)(src +  0), (as3_i4v*)(dstLds +  0), 0, 0);
  __builtin_amdgcn_global_load_async_to_lds_b128((as1_i4v*)(src + 16), (as3_i4v*)(dstLds + 16), 0, 0);
  __builtin_amdgcn_global_load_async_to_lds_b128((as1_i4v*)(src + 32), (as3_i4v*)(dstLds + 32), 0, 0);
  __builtin_amdgcn_global_load_async_to_lds_b128((as1_i4v*)(src + 48), (as3_i4v*)(dstLds + 48), 0, 0);
#else
  const uint4* s = (const uint4*)src;
  uint4 t0 = s[0], t1 = s[1], t2 = s[2], t3 = s[3];
  uint4* d = (uint4*)dstLds;
  d[0] = t0; d[1] = t1; d[2] = t2; d[3] = t3;
#endif
}

__device__ __forceinline__ void stage_wait() {
#if USE_ASYNC_LDS
  __builtin_amdgcn_s_wait_asynccnt(0);
#endif
  __syncthreads();
}

// ---------------- routing ----------------

__global__ void moe_init(int* cnts) {
  if (threadIdx.x < N_EXP) cnts[threadIdx.x] = 0;
}

__global__ __launch_bounds__(256)
void moe_gate(const float* __restrict__ x, const float* __restrict__ Wg,
              int* __restrict__ sel, float* __restrict__ wt, int* __restrict__ cnts)
{
  __shared__ float sWg[D_MODEL * N_EXP];
  for (int i = threadIdx.x; i < D_MODEL * N_EXP; i += 256) sWg[i] = Wg[i];
  __syncthreads();

  int t = blockIdx.x * 256 + threadIdx.x;
  const float* xr = x + (size_t)t * D_MODEL;
  float acc[N_EXP];
#pragma unroll
  for (int e = 0; e < N_EXP; ++e) acc[e] = 0.0f;
  for (int d = 0; d < D_MODEL; d += 4) {
    float4 xv = *(const float4*)(xr + d);
#pragma unroll
    for (int e = 0; e < N_EXP; ++e) {
      acc[e] += xv.x * sWg[(d + 0) * N_EXP + e];
      acc[e] += xv.y * sWg[(d + 1) * N_EXP + e];
      acc[e] += xv.z * sWg[(d + 2) * N_EXP + e];
      acc[e] += xv.w * sWg[(d + 3) * N_EXP + e];
    }
  }
  int e0 = 0; float l0 = acc[0];
#pragma unroll
  for (int e = 1; e < N_EXP; ++e) if (acc[e] > l0) { l0 = acc[e]; e0 = e; }
  int e1 = (e0 == 0) ? 1 : 0; float l1 = acc[e1];
#pragma unroll
  for (int e = 0; e < N_EXP; ++e)
    if (e != e0 && acc[e] > l1) { l1 = acc[e]; e1 = e; }
  float w0 = 1.0f / (1.0f + __expf(l1 - l0));   // softmax over top-2, l0 >= l1
  sel[2 * t]     = e0;  sel[2 * t + 1] = e1;
  wt [2 * t]     = w0;  wt [2 * t + 1] = 1.0f - w0;
  atomicAdd(&cnts[e0], 1);
  atomicAdd(&cnts[e1], 1);
}

__global__ void moe_scan(const int* __restrict__ cnts, int* __restrict__ offs,
                         int* __restrict__ curs)
{
  if (threadIdx.x == 0) {
    int run = 0;
    for (int e = 0; e < N_EXP; ++e) { offs[e] = run; curs[e] = run; run += cnts[e]; }
  }
}

__global__ __launch_bounds__(256)
void moe_scatter(const int* __restrict__ sel, int* __restrict__ curs,
                 int* __restrict__ arow, int* __restrict__ slot)
{
  int t = blockIdx.x * 256 + threadIdx.x;
#pragma unroll
  for (int k = 0; k < TOPK; ++k) {
    int e = sel[2 * t + k];
    int r = atomicAdd(&curs[e], 1);
    arow[r] = t;
    slot[2 * t + k] = r;
  }
}

// ---------------- precision conversion / layout ----------------

__global__ __launch_bounds__(256)
void cvt_f32_bf16_vec4(const float* __restrict__ in, unsigned short* __restrict__ out)
{
  int i = blockIdx.x * 256 + threadIdx.x;
  float4 v = ((const float4*)in)[i];
  uint2 p;
  p.x = (unsigned)f2bf(v.x) | ((unsigned)f2bf(v.y) << 16);
  p.y = (unsigned)f2bf(v.z) | ((unsigned)f2bf(v.w) << 16);
  ((uint2*)out)[i] = p;
}

// fp32 (Rm x Cm) -> bf16 (Cm x Rm), one expert per blockIdx.z
__global__ __launch_bounds__(256)
void transp_cvt(const float* __restrict__ in, unsigned short* __restrict__ out,
                int Rm, int Cm)
{
  __shared__ float tile[32][33];
  const float* inp     = in  + (size_t)blockIdx.z * (size_t)Rm * (size_t)Cm;
  unsigned short* outp = out + (size_t)blockIdx.z * (size_t)Rm * (size_t)Cm;
  int c0 = blockIdx.x * 32, r0 = blockIdx.y * 32;
  int tx = threadIdx.x, ty = threadIdx.y;   // block (32, 8)
#pragma unroll
  for (int i = 0; i < 4; ++i)
    tile[ty + i * 8][tx] = inp[(size_t)(r0 + ty + i * 8) * Cm + (c0 + tx)];
  __syncthreads();
#pragma unroll
  for (int i = 0; i < 4; ++i)
    outp[(size_t)(c0 + ty + i * 8) * Rm + (r0 + tx)] = f2bf(tile[tx][ty + i * 8]);
}

// ---------------- WMMA GEMM: C[m,n] = sum_k A[m,k] * Bt[n,k] ----------------
// FUSE1: A rows gathered via arow, epilogue = +b1, ReLU, bf16 store to h.
// else : A rows are h rows (identity), epilogue = fp32 store to out_buf.

template<bool FUSE1>
__global__ __launch_bounds__(128)
void moe_gemm(const unsigned short* __restrict__ A,
              const unsigned short* __restrict__ Bt,
              const float* __restrict__ bias,
              unsigned short* __restrict__ hout,
              float* __restrict__ fout,
              const int* __restrict__ cnts,
              const int* __restrict__ offs,
              const int* __restrict__ arow,
              int Kd, int Nd)
{
  const int e   = blockIdx.z;
  const int cnt = cnts[e];
  const int m0  = blockIdx.y * 128;
  if (m0 >= cnt) return;
  const int base = offs[e];
  const int n0   = blockIdx.x * 128;
  const int tid  = threadIdx.x;
  const int wv   = tid >> 5, lane = tid & 31;
  const int half = lane >> 4, l16 = lane & 15;

  // double-buffered B tile: 128 rows (N) x 32 bf16 (K), 80B pitch (16B aligned, bank-spread)
  __shared__ unsigned char sB[2][128 * 80];

  const unsigned short* Be = Bt + (size_t)e * (size_t)Nd * (size_t)Kd;

  // per-lane A row pointers for the two 16-row M blocks (ISA 7.12.2: A lane M = lane%16)
  const unsigned char* rowA[2];
#pragma unroll
  for (int b = 0; b < 2; ++b) {
    int mloc = m0 + wv * 32 + b * 16 + l16;
    int mc   = (mloc < cnt) ? mloc : (cnt - 1);
    int grow = FUSE1 ? arow[base + mc] : (base + mc);
    rowA[b] = (const unsigned char*)(A + (size_t)grow * (size_t)Kd);
  }

  v8f acc[2][8];
#pragma unroll
  for (int b = 0; b < 2; ++b)
#pragma unroll
    for (int n = 0; n < 8; ++n)
#pragma unroll
      for (int i = 0; i < 8; ++i) acc[b][n][i] = 0.0f;

  const int nK = Kd >> 5;
  const unsigned char* bsrc = (const unsigned char*)(Be + (size_t)(n0 + tid) * (size_t)Kd);

  // prologue: stage k-step 0 into buffer 0
  stage_b(bsrc, &sB[0][tid * 80]);
  stage_wait();

  for (int kt = 0; kt < nK; ++kt) {
    const int k0 = kt << 5;
    const unsigned char* cur = sB[kt & 1];

    // stage next k-step into the other buffer; overlaps with the 16 WMMAs below
    if (kt + 1 < nK)
      stage_b(bsrc + ((kt + 1) << 6), &sB[(kt + 1) & 1][tid * 80]);

    BfFrag af[2];
#pragma unroll
    for (int b = 0; b < 2; ++b) {   // A frag: two b128 per lane, half selects K chunk
      const unsigned char* p = rowA[b] + 2 * k0 + 16 * half;
      af[b].lo = *(const uint4*)p;
      af[b].hi = *(const uint4*)(p + 32);
    }
    v16bf av0 = __builtin_bit_cast(v16bf, af[0]);
    v16bf av1 = __builtin_bit_cast(v16bf, af[1]);

#pragma unroll
    for (int n = 0; n < 8; ++n) {
      // B frag: lane N = lane%16 -> LDS row n*16+l16; K chunk by half (2x ds_load_b128)
      const unsigned char* q = cur + (n * 16 + l16) * 80 + 16 * half;
      BfFrag bf;
      bf.lo = *(const uint4*)q;
      bf.hi = *(const uint4*)(q + 32);
      v16bf bv = __builtin_bit_cast(v16bf, bf);
      acc[0][n] = __builtin_amdgcn_wmma_f32_16x16x32_bf16(
          false, av0, false, bv, (short)0, acc[0][n], false, false);
      acc[1][n] = __builtin_amdgcn_wmma_f32_16x16x32_bf16(
          false, av1, false, bv, (short)0, acc[1][n], false, false);
    }

    // make next buffer visible (and protect it from the overwrite two steps ahead)
    if (kt + 1 < nK) stage_wait();
  }

  // epilogue: D layout (ISA 7.12.2): lane N = lane%16, VGPR r -> M = r + 8*(lane/16)
#pragma unroll
  for (int b = 0; b < 2; ++b) {
#pragma unroll
    for (int n = 0; n < 8; ++n) {
      const int col = n0 + n * 16 + l16;
      float bv = FUSE1 ? bias[e * Nd + col] : 0.0f;
#pragma unroll
      for (int r = 0; r < 8; ++r) {
        int mloc = m0 + wv * 32 + b * 16 + half * 8 + r;
        if (mloc < cnt) {
          size_t idx = (size_t)(base + mloc) * (size_t)Nd + col;
          float v = acc[b][n][r];
          if (FUSE1) {
            v += bv;
            v = v > 0.0f ? v : 0.0f;
            hout[idx] = f2bf(v);
          } else {
            fout[idx] = v;
          }
        }
      }
    }
  }
}

// ---------------- combine ----------------

__global__ __launch_bounds__(256)
void moe_combine(const float* __restrict__ obuf, const float* __restrict__ b2,
                 const int* __restrict__ sel, const float* __restrict__ wt,
                 const int* __restrict__ slot, float* __restrict__ out)
{
  int idx = blockIdx.x * 256 + threadIdx.x;
  int t = idx >> 10, d = idx & (D_MODEL - 1);
  int r0 = slot[2 * t], r1 = slot[2 * t + 1];
  int e0 = sel[2 * t],  e1 = sel[2 * t + 1];
  float w0 = wt[2 * t], w1 = wt[2 * t + 1];
  float v0 = obuf[(size_t)r0 * D_MODEL + d] + b2[e0 * D_MODEL + d];
  float v1 = obuf[(size_t)r1 * D_MODEL + d] + b2[e1 * D_MODEL + d];
  out[idx] = w0 * v0 + w1 * v1;
}

// ---------------- host ----------------

extern "C" void kernel_launch(void* const* d_in, const int* in_sizes, int n_in,
                              void* d_out, int out_size, void* d_ws, size_t ws_size,
                              hipStream_t stream)
{
  const float* x  = (const float*)d_in[0];
  const float* Wg = (const float*)d_in[1];
  const float* W1 = (const float*)d_in[2];
  const float* b1 = (const float*)d_in[3];
  const float* W2 = (const float*)d_in[4];
  const float* b2 = (const float*)d_in[5];
  float* out = (float*)d_out;
  (void)in_sizes; (void)n_in; (void)out_size; (void)ws_size;

  unsigned char* ws = (unsigned char*)d_ws;
  size_t off = 0;
  auto take = [&](size_t bytes) -> unsigned char* {
    unsigned char* p = ws + off;
    off = (off + bytes + 255) & ~(size_t)255;
    return p;
  };
  unsigned short* xbf  = (unsigned short*)take((size_t)T_TOK * D_MODEL * 2);
  unsigned short* w1t  = (unsigned short*)take((size_t)N_EXP * D_FF * D_MODEL * 2);
  unsigned short* w2t  = (unsigned short*)take((size_t)N_EXP * D_MODEL * D_FF * 2);
  unsigned short* hbuf = (unsigned short*)take((size_t)R_ASN * D_FF * 2);
  float* obuf = (float*)take((size_t)R_ASN * D_MODEL * 4);
  int*   sel  = (int*)take((size_t)T_TOK * 2 * 4);
  float* wt   = (float*)take((size_t)T_TOK * 2 * 4);
  int*   arow = (int*)take((size_t)R_ASN * 4);
  int*   slot = (int*)take((size_t)T_TOK * 2 * 4);
  int*   cnts = (int*)take(64);
  int*   offs = (int*)take(64);
  int*   curs = (int*)take(64);

  moe_init<<<1, 32, 0, stream>>>(cnts);
  moe_gate<<<T_TOK / 256, 256, 0, stream>>>(x, Wg, sel, wt, cnts);
  moe_scan<<<1, 32, 0, stream>>>(cnts, offs, curs);
  moe_scatter<<<T_TOK / 256, 256, 0, stream>>>(sel, curs, arow, slot);

  cvt_f32_bf16_vec4<<<(T_TOK * D_MODEL / 4) / 256, 256, 0, stream>>>(x, xbf);
  transp_cvt<<<dim3(D_FF / 32, D_MODEL / 32, N_EXP), dim3(32, 8), 0, stream>>>(W1, w1t, D_MODEL, D_FF);
  transp_cvt<<<dim3(D_MODEL / 32, D_FF / 32, N_EXP), dim3(32, 8), 0, stream>>>(W2, w2t, D_FF, D_MODEL);

  moe_gemm<true><<<dim3(D_FF / 128, T_TOK / 128, N_EXP), 128, 0, stream>>>(
      xbf, w1t, b1, hbuf, nullptr, cnts, offs, arow, D_MODEL, D_FF);
  moe_gemm<false><<<dim3(D_MODEL / 128, T_TOK / 128, N_EXP), 128, 0, stream>>>(
      hbuf, w2t, nullptr, nullptr, obuf, cnts, offs, arow, D_FF, D_MODEL);

  moe_combine<<<(T_TOK * D_MODEL) / 256, 256, 0, stream>>>(obuf, b2, sel, wt, slot, out);
}
